// GeodesicLayer_22582938042986
// MI455X (gfx1250) — matrix operand order repaired and checked by reading, not compiled, not run on hardware
//
#include <hip/hip_runtime.h>

typedef __attribute__((ext_vector_type(16))) _Float16 v16h;
typedef __attribute__((ext_vector_type(8)))  _Float16 v8h;
typedef __attribute__((ext_vector_type(8)))  float    v8f;

#define P_BINS   5
#define T_BINS   16
#define B_BINS   80            // P_BINS * T_BINS
#define IN_CH    16
#define OUT_CH   32
#define K_NNZ    3
#define K_DIM    (B_BINS * IN_CH)     // 1280
#define N_COLS   (OUT_CH * T_BINS)    // 512
#define K_STEPS  (K_DIM / 32)         // 40
#define M_TILE   32                   // patches per workgroup (2 WMMA M-tiles)
#define LDS_STRIDE (K_DIM + 8)        // pad: 1288 halves -> rows shift 4 banks

// ---------------------------------------------------------------------------
// Kernel 1: pack rotated weights into f16, column-major per output column:
//   lwT[col][k],  col = out*16 + t  (0..511),  k = b*16 + inp (0..1279)
//   value = weights[(b + 5*t) % 80][inp][out]
// ---------------------------------------------------------------------------
__global__ void pack_lw_kernel(const float* __restrict__ weights,
                               _Float16* __restrict__ lwT) {
    int e = blockIdx.x * blockDim.x + threadIdx.x;
    if (e >= N_COLS * K_DIM) return;
    int col = e / K_DIM;
    int k   = e - col * K_DIM;
    int out = col / T_BINS;
    int t   = col - out * T_BINS;
    int b   = k / IN_CH;
    int inp = k - b * IN_CH;
    int src = ((b + P_BINS * t) % B_BINS) * (IN_CH * OUT_CH) + inp * OUT_CH + out;
    lwT[e] = (_Float16)weights[src];
}

// A fragment (16-bit A 16x32 ISA layout): lane holds row M = lane&15,
// elements 0-7 = K half*8.., elements 8-15 = K 16+half*8..
__device__ __forceinline__ v16h load_afrag(const _Float16* base) {
    v8h lo = *(const v8h*)(base);
    v8h hi = *(const v8h*)(base + 16);
    return __builtin_shufflevector(lo, hi,
             0,1,2,3,4,5,6,7,8,9,10,11,12,13,14,15);
}

// B fragment: lane holds column N = lane&15, elements 0-7 = K half*8..,
// elements 8-15 = K 16+half*8.. (matches A fragment).
__device__ __forceinline__ v16h load_bfrag(const _Float16* base) {
    v8h lo = *(const v8h*)(base);
    v8h hi = *(const v8h*)(base + 16);
    return __builtin_shufflevector(lo, hi,
             0,1,2,3,4,5,6,7,8,9,10,11,12,13,14,15);
}

__device__ __forceinline__ v8f wmma_step(v16h a, v16h b, v8f c) {
    return __builtin_amdgcn_wmma_f32_16x16x32_f16(false, a, false, b,
                                                  (short)0, c, false, false);
}

// ---------------------------------------------------------------------------
// Kernel 2: fused gather -> f16 LDS A tile (32 x 1280) -> software-pipelined
// WMMA GEMM (8 accumulators / wave) -> max-over-t epilogue.
// 256 threads = 8 waves; wave w owns out-channels [4w, 4w+4) x 2 M-tiles.
// ---------------------------------------------------------------------------
__global__ __launch_bounds__(256)
void geodesic_fused_kernel(const float* __restrict__ x,
                           const int*   __restrict__ conn_idx,
                           const float* __restrict__ conn_val,
                           const _Float16* __restrict__ lwT,
                           float* __restrict__ out,
                           int n_patch) {
    __shared__ _Float16 Alds[M_TILE * LDS_STRIDE];   // 82,432 bytes

    const int tid    = threadIdx.x;
    const int patch0 = blockIdx.x * M_TILE;
    if (patch0 >= n_patch) return;

    // ---- Phase 1: gather 32 patches x 80 bins x 16 ch into LDS (f16) ----
    // 2560 rows / 256 threads = 10 rows per thread.
    #pragma unroll
    for (int it = 0; it < (M_TILE * B_BINS) / 256; ++it) {
        int r = tid + it * 256;            // 0..2559
        int p = r / B_BINS;                // local patch 0..31
        int b = r - p * B_BINS;            // bin 0..79
        _Float16* dst = &Alds[p * LDS_STRIDE + b * IN_CH];   // 16B aligned

        if (patch0 + p < n_patch) {
            long grow = (long)(patch0 + p) * B_BINS + b;
            const int*   ip = conn_idx + grow * K_NNZ;
            const float* vp = conn_val + grow * K_NNZ;
            int   i0 = ip[0], i1 = ip[1], i2 = ip[2];
            float v0 = vp[0], v1 = vp[1], v2 = vp[2];
            const float4* x0 = (const float4*)(x + (size_t)i0 * IN_CH);
            const float4* x1 = (const float4*)(x + (size_t)i1 * IN_CH);
            const float4* x2 = (const float4*)(x + (size_t)i2 * IN_CH);

            float racc[16];
            #pragma unroll
            for (int q = 0; q < 4; ++q) {
                float4 a = x0[q], c = x1[q], d = x2[q];
                racc[q*4 + 0] = v0 * a.x + v1 * c.x + v2 * d.x;
                racc[q*4 + 1] = v0 * a.y + v1 * c.y + v2 * d.y;
                racc[q*4 + 2] = v0 * a.z + v1 * c.z + v2 * d.z;
                racc[q*4 + 3] = v0 * a.w + v1 * c.w + v2 * d.w;
            }
            v8h h0, h1;
            #pragma unroll
            for (int i = 0; i < 8; ++i) { h0[i] = (_Float16)racc[i]; h1[i] = (_Float16)racc[8 + i]; }
            *(v8h*)(dst)     = h0;
            *(v8h*)(dst + 8) = h1;
        } else {
            v8h z = (v8h)(_Float16)0.0f;
            *(v8h*)(dst)     = z;
            *(v8h*)(dst + 8) = z;
        }
    }
    __syncthreads();

    // ---- Phase 2: software-pipelined WMMA GEMM over K=1280 ----
    const int lane = tid & 31;
    const int wave = tid >> 5;
    const int row  = lane & 15;    // A: M row / B,D: column N
    const int half = lane >> 4;    // lane group select

    v8f acc[2][4];
    #pragma unroll
    for (int m = 0; m < 2; ++m)
        #pragma unroll
        for (int t = 0; t < 4; ++t)
            acc[m][t] = (v8f){0,0,0,0,0,0,0,0};

    const _Float16* Abase0 = &Alds[(0  + row) * LDS_STRIDE + half * 8];
    const _Float16* Abase1 = &Alds[(16 + row) * LDS_STRIDE + half * 8];
    const _Float16* Bbase[4];
    #pragma unroll
    for (int t = 0; t < 4; ++t)
        Bbase[t] = lwT + (size_t)((wave * 4 + t) * T_BINS + row) * K_DIM + half * 8;

    // prime the pipeline (K-step 0)
    v16h aC0 = load_afrag(Abase0);
    v16h aC1 = load_afrag(Abase1);
    v16h bC0 = load_bfrag(Bbase[0]);
    v16h bC1 = load_bfrag(Bbase[1]);
    v16h bC2 = load_bfrag(Bbase[2]);
    v16h bC3 = load_bfrag(Bbase[3]);

    #pragma unroll 2
    for (int ks = 0; ks < K_STEPS; ++ks) {
        // branchless next-step offset (tail re-reads step 0; results unused)
        const int K1 = (ks + 1 < K_STEPS) ? (ks + 1) * 32 : 0;

        // issue next-iteration loads into fresh registers first
        v16h bN0 = load_bfrag(Bbase[0] + K1);
        v16h bN1 = load_bfrag(Bbase[1] + K1);
        v16h bN2 = load_bfrag(Bbase[2] + K1);
        v16h bN3 = load_bfrag(Bbase[3] + K1);
        v16h aN0 = load_afrag(Abase0 + K1);
        v16h aN1 = load_afrag(Abase1 + K1);

        // 8 WMMAs on the already-resident current fragments
        acc[0][0] = wmma_step(aC0, bC0, acc[0][0]);
        acc[1][0] = wmma_step(aC1, bC0, acc[1][0]);
        acc[0][1] = wmma_step(aC0, bC1, acc[0][1]);
        acc[1][1] = wmma_step(aC1, bC1, acc[1][1]);
        acc[0][2] = wmma_step(aC0, bC2, acc[0][2]);
        acc[1][2] = wmma_step(aC1, bC2, acc[1][2]);
        acc[0][3] = wmma_step(aC0, bC3, acc[0][3]);
        acc[1][3] = wmma_step(aC1, bC3, acc[1][3]);

        aC0 = aN0; aC1 = aN1;
        bC0 = bN0; bC1 = bN1; bC2 = bN2; bC3 = bN3;
    }

    // ---- Phase 3: max over the 16 t-bins (= 16 N lanes of each half) ----
    // D layout: VGPR v, lanes 0-15 -> M=v, N=lane; lanes 16-31 -> M=v+8.
    #pragma unroll
    for (int m = 0; m < 2; ++m) {
        #pragma unroll
        for (int t4 = 0; t4 < 4; ++t4) {
            int outCh = wave * 4 + t4;
            #pragma unroll
            for (int v = 0; v < 8; ++v) {
                float mx = acc[m][t4][v];
                mx = fmaxf(mx, __shfl_xor(mx, 8, 16));
                mx = fmaxf(mx, __shfl_xor(mx, 4, 16));
                mx = fmaxf(mx, __shfl_xor(mx, 2, 16));
                mx = fmaxf(mx, __shfl_xor(mx, 1, 16));
                if ((lane & 15) == 0) {
                    int patch = patch0 + m * 16 + v + half * 8;
                    if (patch < n_patch)
                        out[(size_t)patch * OUT_CH + outCh] = mx;
                }
            }
        }
    }
}

// ---------------------------------------------------------------------------
// Host launcher. Inputs (setup_inputs order): x (f32), conn_idx (i32),
// conn_val (f32), weights (f32). Output: (n_patch, 32) f32.
// d_ws holds the packed f16 lw^T (512*1280*2 = 1.31 MB).
// ---------------------------------------------------------------------------
extern "C" void kernel_launch(void* const* d_in, const int* in_sizes, int n_in,
                              void* d_out, int out_size, void* d_ws, size_t ws_size,
                              hipStream_t stream) {
    const float* x        = (const float*)d_in[0];
    const int*   conn_idx = (const int*)  d_in[1];
    const float* conn_val = (const float*)d_in[2];
    const float* weights  = (const float*)d_in[3];
    float*       out      = (float*)d_out;
    _Float16*    lwT      = (_Float16*)d_ws;

    const int n_patch = in_sizes[1] / (B_BINS * K_NNZ);

    const int packN = N_COLS * K_DIM;
    pack_lw_kernel<<<(packN + 255) / 256, 256, 0, stream>>>(weights, lwT);

    const int n_tiles = (n_patch + M_TILE - 1) / M_TILE;
    geodesic_fused_kernel<<<n_tiles, 256, 0, stream>>>(x, conn_idx, conn_val,
                                                       lwT, out, n_patch);
}
